// PointAttN_62088047231548
// MI455X (gfx1250) — matrix operand order repaired and checked by reading, not compiled, not run on hardware
//
#include <hip/hip_runtime.h>
#include <hip/hip_bf16.h>

typedef __attribute__((ext_vector_type(16))) _Float16 v16h;
typedef __attribute__((ext_vector_type(8)))  _Float16 v8h;
typedef __attribute__((ext_vector_type(8)))  float    v8f;
typedef _Float16 half_t;

// ---------------------------------------------------------------------------
// f32 -> f16 conversion
// ---------------------------------------------------------------------------
__global__ void k_f2h(const float* __restrict__ x, half_t* __restrict__ y, long n) {
    long i = (long)blockIdx.x * blockDim.x + threadIdx.x;
    if (i < n) y[i] = (half_t)x[i];
}

// ---------------------------------------------------------------------------
// WMMA GEMM: C[M,N] (f32) = alpha * A[M,K] (f16) x op(B) (f16)
//   TRANSB=1: B stored [N,K] row-major (x @ W^T and Q @ K^T)
//   TRANSB=0: B stored [K,N] row-major (P @ V)
// 128 threads = 4 waves in a 2x2 grid; block tile 64x64; K-steps of 32.
// A tile staged in LDS in WMMA-fragment K-permuted order so every lane reads
// its whole v16h fragment as two aligned ds_load_b128; B staged column-striped
// (Bs[n][k]). Edge paths are branchless (clamped address + select-zero).
// ---------------------------------------------------------------------------
#define BM 64
#define BN 64
#define BK 32
#define LROW 40   // halves per LDS row (80 B stride: 16B-aligned, bank-spread)

template <int TRANSB>
__global__ void __launch_bounds__(128)
k_gemm_wmma(const half_t* __restrict__ A, const half_t* __restrict__ B,
            float* __restrict__ C, int M, int N, int K,
            int lda, int ldb, int ldc, float alpha) {
    __shared__ half_t As[BM * LROW];
    __shared__ half_t Bs[BN * LROW];

    const int tid  = threadIdx.x;
    const int wave = tid >> 5;
    const int lane = tid & 31;
    const int wr   = wave >> 1;   // 0..1
    const int wc   = wave & 1;    // 0..1
    const int m0   = blockIdx.y * BM;
    const int n0   = blockIdx.x * BN;
    const int grp  = lane >> 4;
    const int l16  = lane & 15;

    v8f acc[2][2] = {};

    for (int k0 = 0; k0 < K; k0 += BK) {
        // ---- cooperative tile loads: 2 chunks of 8 halves per thread/tile ----
        #pragma unroll
        for (int cc = 0; cc < 2; ++cc) {
            const int ci = tid * 2 + cc;              // 0..255
            // A tile: BM x BK ; r = row, c8 = k-chunk base
            {
                int r  = ci >> 2;
                int c8 = (ci & 3) << 3;
                int gm = m0 + r, gk = k0 + c8;
                // fragment-order permutation (swaps k[8..15] <-> k[16..23])
                int pc = (c8 & 7) | ((c8 >> 1) & 8) | ((c8 << 1) & 16);
                v8h va;
                if (gm < M && gk + 7 < K) {
                    va = *(const v8h*)(A + (long)gm * lda + gk);
                } else {                               // branchless edge path
                    const long rb = (long)(gm < M ? gm : 0) * lda;
                    #pragma unroll
                    for (int j = 0; j < 8; ++j) {
                        int  kk = gk + j;
                        bool ok = (gm < M) & (kk < K);
                        half_t t = A[rb + (ok ? kk : 0)];
                        va[j] = ok ? t : (half_t)0.f;
                    }
                }
                *(v8h*)(As + r * LROW + pc) = va;
            }
            // B tile -> Bs[n][k]
            if (TRANSB) {
                int r  = ci >> 2;                     // n index 0..63
                int c8 = (ci & 3) << 3;               // k chunk
                int gn = n0 + r, gk = k0 + c8;
                v8h vb;
                if (gn < N && gk + 7 < K) {
                    vb = *(const v8h*)(B + (long)gn * ldb + gk);
                } else {
                    const long rb = (long)(gn < N ? gn : 0) * ldb;
                    #pragma unroll
                    for (int j = 0; j < 8; ++j) {
                        int  kk = gk + j;
                        bool ok = (gn < N) & (kk < K);
                        half_t t = B[rb + (ok ? kk : 0)];
                        vb[j] = ok ? t : (half_t)0.f;
                    }
                }
                *(v8h*)(Bs + r * LROW + c8) = vb;
            } else {
                int kk = ci >> 3;                     // k index 0..31
                int n8 = (ci & 7) << 3;               // n chunk
                int gk = k0 + kk, gn = n0 + n8;
                v8h vb;
                if (gk < K && gn + 7 < N) {
                    vb = *(const v8h*)(B + (long)gk * ldb + gn);
                } else {
                    const long rb = (long)(gk < K ? gk : 0) * ldb;
                    #pragma unroll
                    for (int j = 0; j < 8; ++j) {
                        int  nn = gn + j;
                        bool ok = (gk < K) & (nn < N);
                        half_t t = B[rb + (ok ? nn : 0)];
                        vb[j] = ok ? t : (half_t)0.f;
                    }
                }
                #pragma unroll
                for (int j = 0; j < 8; ++j) Bs[(n8 + j) * LROW + kk] = vb[j];
            }
        }
        // prefetch next A K-tile (speculative, per-lane)
        {
            int pr = m0 + (tid >> 1); if (pr >= M) pr = M - 1;
            long pk = (k0 + BK < K) ? (long)(k0 + BK) : (long)k0;
            __builtin_prefetch(&A[(long)pr * lda + pk], 0, 1);
        }
        __syncthreads();

        // ---- fragments: each is two aligned 16B LDS loads ----
        v16h af[2], bf[2];
        #pragma unroll
        for (int i = 0; i < 2; ++i) {
            int am = wr * 32 + i * 16 + l16;
            v8h a0 = *(const v8h*)(As + am * LROW + grp * 16);
            v8h a1 = *(const v8h*)(As + am * LROW + grp * 16 + 8);
            af[i] = __builtin_shufflevector(a0, a1, 0, 1, 2, 3, 4, 5, 6, 7,
                                            8, 9, 10, 11, 12, 13, 14, 15);
        }
        #pragma unroll
        for (int j = 0; j < 2; ++j) {
            int bn = wc * 32 + j * 16 + l16;
            v8h b0 = *(const v8h*)(Bs + bn * LROW + grp * 8);
            v8h b1 = *(const v8h*)(Bs + bn * LROW + 16 + grp * 8);
            bf[j] = __builtin_shufflevector(b0, b1, 0, 1, 2, 3, 4, 5, 6, 7,
                                            8, 9, 10, 11, 12, 13, 14, 15);
        }
        #pragma unroll
        for (int i = 0; i < 2; ++i)
            #pragma unroll
            for (int j = 0; j < 2; ++j)
                acc[i][j] = __builtin_amdgcn_wmma_f32_16x16x32_f16(
                    false, af[i], false, bf[j], (short)0, acc[i][j], false, false);
        __syncthreads();
    }

    // ---- C store: VGPR v -> row v + 8*grp, col = lane&15 (ISA 7.12.2) ----
    if (m0 + BM <= M && n0 + BN <= N) {           // fast path: unconditional
        #pragma unroll
        for (int i = 0; i < 2; ++i) {
            #pragma unroll
            for (int j = 0; j < 2; ++j) {
                int cn = n0 + wc * 32 + j * 16 + l16;
                #pragma unroll
                for (int v = 0; v < 8; ++v) {
                    int cm = m0 + wr * 32 + i * 16 + v + 8 * grp;
                    C[(long)cm * ldc + cn] = alpha * acc[i][j][v];
                }
            }
        }
    } else {                                      // edge blocks only
        #pragma unroll
        for (int i = 0; i < 2; ++i) {
            #pragma unroll
            for (int j = 0; j < 2; ++j) {
                int cn = n0 + wc * 32 + j * 16 + l16;
                #pragma unroll
                for (int v = 0; v < 8; ++v) {
                    int cm = m0 + wr * 32 + i * 16 + v + 8 * grp;
                    if (cm < M && cn < N) C[(long)cm * ldc + cn] = alpha * acc[i][j][v];
                }
            }
        }
    }
}

// ---------------------------------------------------------------------------
// bias (+ optional ReLU) epilogue, in place
// ---------------------------------------------------------------------------
__global__ void k_bias_act(float* __restrict__ y, const float* __restrict__ b,
                           long R, int N, int relu) {
    long i = (long)blockIdx.x * blockDim.x + threadIdx.x;
    if (i >= R * (long)N) return;
    int n = (int)(i % N);
    float v = y[i] + b[n];
    if (relu) v = fmaxf(v, 0.f);
    y[i] = v;
}

// ---------------------------------------------------------------------------
// row LayerNorm (block per row); x may alias y
// ---------------------------------------------------------------------------
__global__ void __launch_bounds__(256)
k_layernorm(const float* __restrict__ x, float* __restrict__ y,
            const float* __restrict__ g, const float* __restrict__ be, int D) {
    __shared__ float red[256];
    const float* xr = x + (long)blockIdx.x * D;
    float s = 0.f;
    for (int d = threadIdx.x; d < D; d += 256) s += xr[d];
    red[threadIdx.x] = s; __syncthreads();
    for (int o = 128; o > 0; o >>= 1) {
        if (threadIdx.x < o) red[threadIdx.x] += red[threadIdx.x + o];
        __syncthreads();
    }
    float mean = red[0] / D; __syncthreads();
    float v = 0.f;
    for (int d = threadIdx.x; d < D; d += 256) { float t = xr[d] - mean; v += t * t; }
    red[threadIdx.x] = v; __syncthreads();
    for (int o = 128; o > 0; o >>= 1) {
        if (threadIdx.x < o) red[threadIdx.x] += red[threadIdx.x + o];
        __syncthreads();
    }
    float inv = rsqrtf(red[0] / D + 1e-5f);
    float* yr = y + (long)blockIdx.x * D;
    for (int d = threadIdx.x; d < D; d += 256)
        yr[d] = (xr[d] - mean) * inv * g[d] + be[d];
}

// ---------------------------------------------------------------------------
// row softmax, in place (block per row)
// ---------------------------------------------------------------------------
__global__ void __launch_bounds__(256) k_softmax(float* __restrict__ x, int L) {
    __shared__ float red[256];
    float* xr = x + (long)blockIdx.x * L;
    float m = -1e30f;
    for (int i = threadIdx.x; i < L; i += 256) m = fmaxf(m, xr[i]);
    red[threadIdx.x] = m; __syncthreads();
    for (int o = 128; o > 0; o >>= 1) {
        if (threadIdx.x < o) red[threadIdx.x] = fmaxf(red[threadIdx.x], red[threadIdx.x + o]);
        __syncthreads();
    }
    m = red[0]; __syncthreads();
    float s = 0.f;
    for (int i = threadIdx.x; i < L; i += 256) { float e = __expf(xr[i] - m); xr[i] = e; s += e; }
    red[threadIdx.x] = s; __syncthreads();
    for (int o = 128; o > 0; o >>= 1) {
        if (threadIdx.x < o) red[threadIdx.x] += red[threadIdx.x + o];
        __syncthreads();
    }
    float inv = 1.f / red[0];
    for (int i = threadIdx.x; i < L; i += 256) xr[i] *= inv;
}

// ---------------------------------------------------------------------------
// farthest point sampling; one block per batch; dist kept in LDS.
// Matches reference scan semantics: record 'far' then update.
// ---------------------------------------------------------------------------
__global__ void __launch_bounds__(256)
k_fps(const float* __restrict__ pts, int Ntot, int N, int k, int* __restrict__ idx) {
    __shared__ float dist[4096];
    __shared__ float rv[256];
    __shared__ int   ri[256];
    __shared__ int   s_far;
    const int b = blockIdx.x;
    const float* cb = pts + (long)b * Ntot * 3;
    for (int i = threadIdx.x; i < N; i += 256) dist[i] = 1e10f;
    if (threadIdx.x == 0) s_far = 0;
    __syncthreads();
    for (int it = 0; it < k; ++it) {
        int far = s_far;
        if (threadIdx.x == 0) idx[b * k + it] = far;
        float cx = cb[far * 3 + 0], cy = cb[far * 3 + 1], cz = cb[far * 3 + 2];
        float bm = -1.f; int bi = 0;
        for (int i = threadIdx.x; i < N; i += 256) {
            float dx = cb[i * 3 + 0] - cx, dy = cb[i * 3 + 1] - cy, dz = cb[i * 3 + 2] - cz;
            float d = dx * dx + dy * dy + dz * dz;
            float nd = fminf(dist[i], d);
            dist[i] = nd;
            if (nd > bm) { bm = nd; bi = i; }
        }
        rv[threadIdx.x] = bm; ri[threadIdx.x] = bi; __syncthreads();
        for (int o = 128; o > 0; o >>= 1) {
            if (threadIdx.x < o) {
                if (rv[threadIdx.x + o] > rv[threadIdx.x] ||
                    (rv[threadIdx.x + o] == rv[threadIdx.x] && ri[threadIdx.x + o] < ri[threadIdx.x])) {
                    rv[threadIdx.x] = rv[threadIdx.x + o];
                    ri[threadIdx.x] = ri[threadIdx.x + o];
                }
            }
            __syncthreads();
        }
        if (threadIdx.x == 0) s_far = ri[0];
        __syncthreads();
    }
}

// ---------------------------------------------------------------------------
// gather rows by index: y[b,j,:] = f[b, idx[b,j], :]
// ---------------------------------------------------------------------------
__global__ void k_gather(const float* __restrict__ f, const int* __restrict__ idx,
                         float* __restrict__ y, int Lsrc, int k, int D, long n) {
    long i = (long)blockIdx.x * blockDim.x + threadIdx.x;
    if (i >= n) return;
    int d = (int)(i % D);
    long t = i / D;
    int j = (int)(t % k);
    int b = (int)(t / k);
    y[i] = f[((long)b * Lsrc + idx[b * k + j]) * D + d];
}

__global__ void k_add(const float* __restrict__ a, const float* __restrict__ b,
                      float* __restrict__ y, long n) {
    long i = (long)blockIdx.x * blockDim.x + threadIdx.x;
    if (i < n) y[i] = a[i] + b[i];
}

// max over points: y[b,d] = max_l x[b,l,d]  (D == 256 == blockDim)
__global__ void k_maxpool(const float* __restrict__ x, float* __restrict__ y,
                          int L, int D) {
    int b = blockIdx.x, d = threadIdx.x;
    float m = -1e30f;
    for (int l = 0; l < L; ++l) m = fmaxf(m, x[((long)b * L + l) * D + d]);
    y[b * D + d] = m;
}

// ---------------------------------------------------------------------------
// host orchestration
// ---------------------------------------------------------------------------
extern "C" void kernel_launch(void* const* d_in, const int* in_sizes, int n_in,
                              void* d_out, int out_size, void* d_ws, size_t ws_size,
                              hipStream_t stream) {
    const int D = 256, Bb = 8, N0 = 4096, H = 4, HD = 64;

    // jax pytree flattening sorts dict keys -> params leaves first, pts last.
    // Fallback: if pts (size 98304, shape [8,4096,3]) is first, shift.
    int p0 = 0;
    const float* pts;
    if (in_sizes[0] == Bb * N0 * 3) { pts = (const float*)d_in[0]; p0 = 1; }
    else                            { pts = (const float*)d_in[n_in - 1]; p0 = 0; }
    auto P = [&](int i) { return (const float*)d_in[p0 + i]; };
    // param leaf indices (sorted-key order):
    // dec: 0 b1, 1 b2, 2 be1, 3 be2, 4 g1, 5 g2, 6 w1, 7 w2
    // gdp{1,2,3} base 8/14/20: +0 in_b, +1 in_w, +2 nb, +3 ng, +4 out_b, +5 out_w
    // inp base 26: +0 b, +1 be, +2 g, +3 w
    // sfa{1,2,3} base 30/46/62: +0 f1b,+1 f1be,+2 f1g,+3 f1w,+4 f2b,+5 f2be,
    //   +6 f2g,+7 f2w,+8 in_b,+9 in_w,+10 n1b,+11 n1g,+12 n2b,+13 n2g,+14 out_b,+15 out_w

    // -------- workspace arena --------
    char* wsb = (char*)d_ws;
    size_t wo = 0;
    auto alloc = [&](size_t bytes) -> void* {
        void* p = wsb + wo;
        wo += (bytes + 255) & ~(size_t)255;
        return p;
    };
    float*  fe0  = (float*) alloc((size_t)Bb * 4096 * D * 4);   // feats after input MLP
    float*  fA   = (float*) alloc((size_t)Bb * 1024 * D * 4);
    float*  fB   = (float*) alloc((size_t)Bb * 1024 * D * 4);
    float*  kvb  = (float*) alloc((size_t)Bb * 1024 * D * 4);   // gathered key_val
    float*  qf   = (float*) alloc((size_t)Bb * 1024 * D * 4);
    float*  kf   = (float*) alloc((size_t)Bb * 4096 * D * 4);
    float*  vf   = (float*) alloc((size_t)Bb * 4096 * D * 4);
    float*  at   = (float*) alloc((size_t)Bb * 1024 * D * 4);   // attn pre-out-proj
    float*  sc   = (float*) alloc((size_t)1024 * 4096 * 4);     // per (b,h) scores
    float*  hid  = (float*) alloc((size_t)Bb * 1024 * 2 * D * 4);
    half_t* ha   = (half_t*)alloc((size_t)Bb * 4096 * D * 2);   // activation f16
    half_t* hb   = (half_t*)alloc((size_t)12288 * 512 * 2);     // weight f16
    half_t* qh   = (half_t*)alloc((size_t)Bb * 1024 * D * 2);
    half_t* kh   = (half_t*)alloc((size_t)Bb * 4096 * D * 2);
    half_t* vh   = (half_t*)alloc((size_t)Bb * 4096 * D * 2);
    half_t* ph   = (half_t*)alloc((size_t)1024 * 4096 * 2);     // probs f16
    int*    idxb = (int*)   alloc((size_t)Bb * 1024 * 4);
    float*  code = (float*) alloc((size_t)Bb * D * 4);
    float*  dech = (float*) alloc((size_t)Bb * 2 * D * 4);

    // -------- helpers --------
    auto f2h = [&](const float* x, half_t* y, long n) {
        k_f2h<<<(unsigned)((n + 255) / 256), 256, 0, stream>>>(x, y, n);
    };
    auto gemm = [&](const half_t* A, const half_t* Bm, float* C, int M, int N, int K,
                    int lda, int ldb, int ldc, float alpha, bool transB) {
        dim3 g((N + BN - 1) / BN, (M + BM - 1) / BM);
        if (transB) k_gemm_wmma<1><<<g, 128, 0, stream>>>(A, Bm, C, M, N, K, lda, ldb, ldc, alpha);
        else        k_gemm_wmma<0><<<g, 128, 0, stream>>>(A, Bm, C, M, N, K, lda, ldb, ldc, alpha);
    };
    auto biasact = [&](float* y, const float* b, long R, int N, bool relu) {
        long n = R * (long)N;
        k_bias_act<<<(unsigned)((n + 255) / 256), 256, 0, stream>>>(y, b, R, N, relu ? 1 : 0);
    };
    auto linear = [&](const float* x, const float* w, const float* bias, float* y,
                      long R, int K, int Nn, bool relu) {
        f2h(x, ha, R * (long)K);
        f2h(w, hb, (long)Nn * K);
        gemm(ha, hb, y, (int)R, Nn, K, K, K, Nn, 1.f, true);
        biasact(y, bias, R, Nn, relu);
    };
    auto lnorm = [&](const float* x, float* y, const float* g, const float* be,
                     long rows, int Dd) {
        k_layernorm<<<(unsigned)rows, 256, 0, stream>>>(x, y, g, be, Dd);
    };
    auto addv = [&](const float* a, const float* b, float* y, long n) {
        k_add<<<(unsigned)((n + 255) / 256), 256, 0, stream>>>(a, b, y, n);
    };
    auto mha = [&](const float* q_in, const float* kv_in, int Lq, int Lk,
                   const float* in_w, const float* in_b,
                   const float* ow, const float* ob, float* out) {
        linear(q_in,  in_w,                 in_b,         qf, (long)Bb * Lq, D, D, false);
        linear(kv_in, in_w + (long)D * D,   in_b + D,     kf, (long)Bb * Lk, D, D, false);
        linear(kv_in, in_w + 2L * D * D,    in_b + 2 * D, vf, (long)Bb * Lk, D, D, false);
        f2h(qf, qh, (long)Bb * Lq * D);
        f2h(kf, kh, (long)Bb * Lk * D);
        f2h(vf, vh, (long)Bb * Lk * D);
        for (int b = 0; b < Bb; ++b) {
            for (int h = 0; h < H; ++h) {
                const half_t* Aq = qh + ((long)b * Lq * D + h * HD);
                const half_t* Bk = kh + ((long)b * Lk * D + h * HD);
                gemm(Aq, Bk, sc, Lq, Lk, HD, D, D, Lk, 0.125f, true);   // 1/sqrt(64)
                k_softmax<<<Lq, 256, 0, stream>>>(sc, Lk);
                f2h(sc, ph, (long)Lq * Lk);
                const half_t* Bv = vh + ((long)b * Lk * D + h * HD);
                gemm(ph, Bv, at + ((long)b * Lq * D + h * HD),
                     Lq, HD, Lk, Lk, D, D, 1.f, false);
            }
        }
        linear(at, ow, ob, out, (long)Bb * Lq, D, D, false);
    };
    auto gdp = [&](const float* feats, int N, int down,
                   const float* in_w, const float* in_b,
                   const float* ow, const float* ob,
                   const float* ng, const float* nb, float* out) {
        int k = N / down;
        k_fps<<<Bb, 256, 0, stream>>>(pts, N0, N, k, idxb);
        long n = (long)Bb * k * D;
        k_gather<<<(unsigned)((n + 255) / 256), 256, 0, stream>>>(feats, idxb, kvb, N, k, D, n);
        mha(kvb, feats, k, N, in_w, in_b, ow, ob, out);
        addv(out, kvb, out, n);
        lnorm(out, out, ng, nb, (long)Bb * k, D);
    };
    auto sfa = [&](float* x, int L, int si, float* tmp) {
        mha(x, x, L, L, P(si + 9), P(si + 8), P(si + 15), P(si + 14), tmp);
        long n = (long)Bb * L * D;
        addv(tmp, x, tmp, n);
        lnorm(tmp, tmp, P(si + 11), P(si + 10), (long)Bb * L, D);       // n1
        linear(tmp, P(si + 3), P(si + 0), hid, (long)Bb * L, D, 2 * D, true);   // f1
        lnorm(hid, hid, P(si + 2), P(si + 1), (long)Bb * L, 2 * D);
        linear(hid, P(si + 7), P(si + 4), x, (long)Bb * L, 2 * D, D, true);     // f2
        lnorm(x, x, P(si + 6), P(si + 5), (long)Bb * L, D);
        addv(x, tmp, x, n);
        lnorm(x, x, P(si + 13), P(si + 12), (long)Bb * L, D);           // n2
    };

    // -------- forward pass --------
    // input MLP: [B*4096,3] -> [B*4096,256], ReLU, LN
    linear(pts, P(29), P(26), fe0, (long)Bb * N0, 3, D, true);
    lnorm(fe0, fe0, P(28), P(27), (long)Bb * N0, D);

    gdp(fe0, 4096, 4, P(9),  P(8),  P(13), P(12), P(11), P(10), fA);   // -> [B,1024,D]
    sfa(fA, 1024, 30, fB);
    gdp(fA, 1024, 2, P(15), P(14), P(19), P(18), P(17), P(16), fB);    // -> [B,512,D]
    sfa(fB, 512, 46, fA);
    gdp(fB, 512, 2, P(21), P(20), P(25), P(24), P(23), P(22), fA);     // -> [B,256,D]
    sfa(fA, 256, 62, fB);

    k_maxpool<<<Bb, 256, 0, stream>>>(fA, code, 256, D);

    // decoder; final GEMM output is exactly d_out's 8*12288 floats
    linear(code, P(6), P(0), dech, Bb, D, 2 * D, true);
    lnorm(dech, dech, P(4), P(2), Bb, 2 * D);
    float* outp = (float*)d_out;
    linear(dech, P(7), P(1), outp, Bb, 2 * D, 12288, true);
    lnorm(outp, outp, P(5), P(3), Bb, 12288);
}
